// BidirectionalConvMambaBlock_63221918597616
// MI455X (gfx1250) — matrix operand order repaired and checked by reading, not compiled, not run on hardware
//
#include <hip/hip_runtime.h>
#include <math.h>

// ---------------------------------------------------------------------------
// Problem constants (from reference)
// ---------------------------------------------------------------------------
#define DIM     256
#define DSTATE  32
#define DINNER  512
#define DTRANK  16
#define DCONV   4
#define BATCH   8
#define SEQL    2048
#define NTOK    (BATCH * SEQL)          // 16384 tokens
#define RMS_EPS 1.1920929e-07f
#define LN_EPS  1e-05f

typedef __attribute__((ext_vector_type(16))) _Float16 v16h;
typedef __attribute__((ext_vector_type(8)))  float    v8f;

// ---------------------------------------------------------------------------
// Device helpers
// ---------------------------------------------------------------------------
__device__ __forceinline__ float silu_f(float v)  { return v / (1.f + __expf(-v)); }
__device__ __forceinline__ float softplus_f(float v) {
    return fmaxf(v, 0.f) + log1pf(__expf(-fabsf(v)));
}
__device__ __forceinline__ float gelu_f(float v) {
    return 0.5f * v * (1.f + erff(v * 0.70710678118654752f));
}

// 256-thread block sum reduction (wave32 shuffles + 8-partial LDS combine)
__device__ __forceinline__ float block_reduce_sum_256(float v) {
    __shared__ float sred[8];
    #pragma unroll
    for (int o = 16; o; o >>= 1) v += __shfl_xor(v, o, 32);
    __syncthreads();                         // protect sred reuse across calls
    if ((threadIdx.x & 31) == 0) sred[threadIdx.x >> 5] = v;
    __syncthreads();
    float r = 0.f;
    #pragma unroll
    for (int i = 0; i < 8; ++i) r += sred[i];
    return r;
}

// ---------------------------------------------------------------------------
// Generic WMMA GEMM:  C[M x Ncols] = A[M x K] (fp32) @ W[K x Ncols] (fp32)
// fp32 -> f16 staged through double-buffered LDS; fp32 accumulation via
// v_wmma_f32_16x16x32_f16. Block tile 128x64, 8 waves; each wave owns a
// 64-row half x 16-col strip: 4 accumulators -> 32 WMMAs per K-step/block.
// One barrier per K-step (ping-pong buffers). All fragments preloaded, then
// 4 back-to-back WMMAs (single s_wait_dscnt per step).
// NB: column bounds checking (only the Ncols=80 GEMM needs it).
// EPI: 0 = plain store, 1 = +bias then exact GELU, 2 = +bias + residual add.
// M must be a multiple of 128; K a multiple of 32 (true for all uses).
// ---------------------------------------------------------------------------
#define BLK_M 128
#define BLK_N 64
#define BLK_K 32

template <int EPI, bool NB>
__global__ __launch_bounds__(256) void gemm_wmma_f16(
    const float* __restrict__ A, int lda,
    const float* __restrict__ W, int ldw,
    float* __restrict__ C, int ldc,
    int K, int Ncols,
    const float* __restrict__ bias,
    const float* __restrict__ res, int ldres)
{
    __shared__ __align__(16) _Float16 sA[2][BLK_M][BLK_K + 8];  // [m][k], 80B rows
    __shared__ __align__(16) _Float16 sB[2][BLK_N][BLK_K + 8];  // [n][k] (transposed W)

    const int tid  = threadIdx.x;
    const int lane = tid & 31;
    const int wave = tid >> 5;
    const int wn   = wave & 3;     // N sub-tile (x16)
    const int wm   = wave >> 2;    // 0..1 -> 64-row half
    const int half = lane >> 4;    // ISA 16-bit operand layout split
    const int l16  = lane & 15;

    const int rowBase = blockIdx.x * BLK_M;
    const int colBase = blockIdx.y * BLK_N;

    // A staging: 4 float4 per thread. idx4 = tid + 256*i -> m = idx4/8, kc=(idx4%8)*4
    const int kcA = (tid & 7) * 4;
    const int mA0 = tid >> 3;            // + 32*i
    // B staging: 2 float4 per thread. idx4 = tid + 256*i -> n4=(idx4%16)*4, k=idx4/16
    const int n4B = (tid & 15) * 4;
    const int kB0 = tid >> 4;            // + 16*i

    v8f acc[4] = {{}, {}, {}, {}};

    const int KT = K >> 5;
    for (int kt = 0; kt < KT; ++kt) {
        const int k0  = kt << 5;
        const int buf = kt & 1;

        // ---- issue all global loads first (stay in flight) ---------------
        float4 ra[4];
        #pragma unroll
        for (int i = 0; i < 4; ++i)
            ra[i] = *(const float4*)&A[(size_t)(rowBase + mA0 + 32 * i) * lda + k0 + kcA];

        float4 rb[2];
        #pragma unroll
        for (int i = 0; i < 2; ++i) {
            const int col = colBase + n4B;
            const float* wp = &W[(size_t)(k0 + kB0 + 16 * i) * ldw + col];
            if (!NB) {
                rb[i] = *(const float4*)wp;          // branch-free hot path
            } else if (col + 3 < Ncols) {
                rb[i] = *(const float4*)wp;
            } else {
                rb[i].x = (col + 0 < Ncols) ? wp[0] : 0.f;
                rb[i].y = (col + 1 < Ncols) ? wp[1] : 0.f;
                rb[i].z = (col + 2 < Ncols) ? wp[2] : 0.f;
                rb[i].w = (col + 3 < Ncols) ? wp[3] : 0.f;
            }
        }
        // prefetch next K tiles (global_prefetch_b8)
        if (k0 + BLK_K < K) {
            __builtin_prefetch(&A[(size_t)(rowBase + mA0) * lda + k0 + BLK_K + kcA], 0, 1);
            __builtin_prefetch(&W[(size_t)(k0 + BLK_K + kB0) * ldw + colBase + n4B], 0, 1);
        }

        // ---- convert + packed LDS stores into ping-pong buffer -----------
        #pragma unroll
        for (int i = 0; i < 4; ++i) {
            union { _Float16 h[4]; uint2 u; } p;
            p.h[0] = (_Float16)ra[i].x;
            p.h[1] = (_Float16)ra[i].y;
            p.h[2] = (_Float16)ra[i].z;
            p.h[3] = (_Float16)ra[i].w;
            *(uint2*)&sA[buf][mA0 + 32 * i][kcA] = p.u;     // ds_store_b64
        }
        #pragma unroll
        for (int i = 0; i < 2; ++i) {                       // transpose scatter
            sB[buf][n4B + 0][kB0 + 16 * i] = (_Float16)rb[i].x;
            sB[buf][n4B + 1][kB0 + 16 * i] = (_Float16)rb[i].y;
            sB[buf][n4B + 2][kB0 + 16 * i] = (_Float16)rb[i].z;
            sB[buf][n4B + 3][kB0 + 16 * i] = (_Float16)rb[i].w;
        }
        __syncthreads();      // single barrier per K-step (double-buffered)

        // ---- preload ALL fragments, then 4 back-to-back WMMAs ------------
        // B 32x16 ([n][k] LDS): lane n=l16; elems 0-7: k=half*8+j ; elems 8-15: k=16+half*8+j
        v16h b;
        #pragma unroll
        for (int j = 0; j < 8; ++j) {
            b[j]     = sB[buf][wn * 16 + l16][half * 8 + j];
            b[j + 8] = sB[buf][wn * 16 + l16][16 + half * 8 + j];
        }
        // A 16x32: lane m=l16; elems 0-7: k=half*8+j ; elems 8-15: k=16+half*8+j
        v16h a[4];
        #pragma unroll
        for (int s = 0; s < 4; ++s) {
            const int mrow = wm * 64 + s * 16 + l16;
            #pragma unroll
            for (int j = 0; j < 8; ++j) {
                a[s][j]     = sA[buf][mrow][half * 8 + j];
                a[s][j + 8] = sA[buf][mrow][16 + half * 8 + j];
            }
        }
        #pragma unroll
        for (int s = 0; s < 4; ++s)
            acc[s] = __builtin_amdgcn_wmma_f32_16x16x32_f16(false, a[s], false, b,
                                                            (short)0, acc[s], false, false);
    }

    // ---- epilogue + store: C/D layout: elem j -> M = j + 8*half, N = l16 --
    const int col = colBase + wn * 16 + l16;
    if (!NB || col < Ncols) {
        const float bv = (EPI != 0) ? bias[col] : 0.f;
        #pragma unroll
        for (int s = 0; s < 4; ++s) {
            #pragma unroll
            for (int j = 0; j < 8; ++j) {
                const int row = rowBase + wm * 64 + s * 16 + 8 * half + j;
                float v = acc[s][j];
                if (EPI == 1)      v = gelu_f(v + bv);
                else if (EPI == 2) v += bv + res[(size_t)row * ldres + col];
                C[(size_t)row * ldc + col] = v;
            }
        }
    }
}

// ---------------------------------------------------------------------------
// RMSNorm over 256 columns (one block per token row)
// ---------------------------------------------------------------------------
__global__ __launch_bounds__(256) void rmsnorm256_kernel(
    const float* __restrict__ x, const float* __restrict__ w,
    float* __restrict__ out)
{
    const size_t n = blockIdx.x;
    const int d = threadIdx.x;
    float v = x[n * DIM + d];
    float ss = block_reduce_sum_256(v * v);
    float r = rsqrtf(ss * (1.f / DIM) + RMS_EPS);
    out[n * DIM + d] = v * r * w[d];
}

// ---------------------------------------------------------------------------
// Depthwise conv(k=3, same-pad) + LayerNorm + SiLU + add xn  ->  u
// ---------------------------------------------------------------------------
__global__ __launch_bounds__(256) void lconv_ln_silu_kernel(
    const float* __restrict__ xn,
    const float* __restrict__ lw, const float* __restrict__ lb,
    const float* __restrict__ lnw, const float* __restrict__ lnb,
    float* __restrict__ u)
{
    const size_t n = blockIdx.x;
    const int t = (int)(n % SEQL);
    const int d = threadIdx.x;

    float x0 = xn[n * DIM + d];
    float xm = (t > 0)        ? xn[(n - 1) * DIM + d] : 0.f;
    float xp = (t < SEQL - 1) ? xn[(n + 1) * DIM + d] : 0.f;
    float c = lw[d * 3 + 0] * xm + lw[d * 3 + 1] * x0 + lw[d * 3 + 2] * xp + lb[d];

    float mu  = block_reduce_sum_256(c) * (1.f / DIM);
    float dv  = c - mu;
    float var = block_reduce_sum_256(dv * dv) * (1.f / DIM);
    float y = dv * rsqrtf(var + LN_EPS) * lnw[d] + lnb[d];
    u[n * DIM + d] = silu_f(y) + x0;
}

// ---------------------------------------------------------------------------
// Mamba depthwise causal conv (k=4) + SiLU, both directions in one launch.
// fwd: out[t] = sum_k w[k]*xr[t-3+k] ; bwd (reversed pipeline, stored at
// original t): out[t] = sum_k w[k]*xr[t+3-k]. xr = xz[:, :512].
// ---------------------------------------------------------------------------
__global__ __launch_bounds__(256) void mamba_conv_kernel(
    const float* __restrict__ xz,
    const float* __restrict__ cw, const float* __restrict__ cb,
    float* __restrict__ xcf, float* __restrict__ xcb)
{
    size_t tid = (size_t)blockIdx.x * 256 + threadIdx.x;   // 2 * NTOK * DINNER
    const int d = (int)(tid % DINNER);
    size_t r = tid / DINNER;
    const int dir = (int)(r / NTOK);
    const size_t n = r % NTOK;
    const int t = (int)(n % SEQL);

    float acc = cb[d];
    #pragma unroll
    for (int k = 0; k < DCONV; ++k) {
        int p = dir ? (t + (DCONV - 1) - k) : (t - (DCONV - 1) + k);
        if (p >= 0 && p < SEQL)
            acc += cw[d * DCONV + k] * xz[(n - t + p) * (2 * DINNER) + d];
    }
    float v = silu_f(acc);
    (dir ? xcb : xcf)[n * DINNER + d] = v;
}

// ---------------------------------------------------------------------------
// dt = softplus(proj[:, :16] @ dtproj_w + dtproj_b)   (K=16 -> scalar dot)
// ---------------------------------------------------------------------------
__global__ __launch_bounds__(256) void dt_kernel(
    const float* __restrict__ proj,
    const float* __restrict__ dw, const float* __restrict__ db,
    float* __restrict__ dt)
{
    size_t tid = (size_t)blockIdx.x * 256 + threadIdx.x;   // NTOK * DINNER
    const int d = (int)(tid % DINNER);
    const size_t n = tid / DINNER;
    float v = db[d];
    #pragma unroll
    for (int r = 0; r < DTRANK; ++r)
        v += proj[n * 80 + r] * dw[r * DINNER + d];
    dt[n * DINNER + d] = softplus_f(v);
}

// ---------------------------------------------------------------------------
// Selective scan: one wave32 per (b, d); lane = state s. Gated with silu(z).
// y is written in place over xc (read-before-write at the same element).
// ---------------------------------------------------------------------------
__global__ __launch_bounds__(256) void scan_kernel(
    const float* __restrict__ xz,       // z gate = xz[:, 512 + d]
    float* __restrict__ xc,             // in: conv+silu ; out: gated y
    const float* __restrict__ dt,
    const float* __restrict__ proj,     // B = cols 16..47, C = cols 48..79
    const float* __restrict__ A_log, const float* __restrict__ Dm,
    int dir)
{
    const int wid  = blockIdx.x * 8 + (threadIdx.x >> 5);  // 0 .. 4095
    const int lane = threadIdx.x & 31;
    const int b = wid >> 9;
    const int d = wid & 511;

    const float As = -__expf(A_log[d * DSTATE + lane]);
    const float Dd = Dm[d];
    float h = 0.f;

    for (int tt = 0; tt < SEQL; ++tt) {
        const int t = dir ? (SEQL - 1 - tt) : tt;
        const size_t n = (size_t)b * SEQL + t;
        float xt  = xc[n * DINNER + d];
        float dtt = dt[n * DINNER + d];
        float Bs  = proj[n * 80 + DTRANK + lane];
        float Cs  = proj[n * 80 + DTRANK + DSTATE + lane];

        h = __expf(dtt * As) * h + (dtt * xt) * Bs;

        float c = h * Cs;
        #pragma unroll
        for (int o = 16; o; o >>= 1) c += __shfl_xor(c, o, 32);

        if (lane == 0) {
            float y = c + xt * Dd;
            float zv = xz[n * (2 * DINNER) + DINNER + d];
            y *= silu_f(zv);
            xc[n * DINNER + d] = y;
        }
    }
}

// ---------------------------------------------------------------------------
// LayerNorm over 512 cols, in place (each thread owns cols d and d+256)
// ---------------------------------------------------------------------------
__global__ __launch_bounds__(256) void ln512_kernel(
    float* __restrict__ xs, const float* __restrict__ w,
    const float* __restrict__ b)
{
    const size_t n = blockIdx.x;
    const int d = threadIdx.x;
    float v0 = xs[n * 512 + d];
    float v1 = xs[n * 512 + d + 256];
    float mu  = block_reduce_sum_256(v0 + v1) * (1.f / 512.f);
    float d0 = v0 - mu, d1 = v1 - mu;
    float var = block_reduce_sum_256(d0 * d0 + d1 * d1) * (1.f / 512.f);
    float rs = rsqrtf(var + LN_EPS);
    xs[n * 512 + d]       = d0 * rs * w[d] + b[d];
    xs[n * 512 + d + 256] = d1 * rs * w[d + 256] + b[d + 256];
}

// ---------------------------------------------------------------------------
// Post conv: groups=DIM, 2 in-channels per group (xsn cols 2d, 2d+1), k=3,
// same-pad; SiLU; add input residual -> xres
// ---------------------------------------------------------------------------
__global__ __launch_bounds__(256) void pconv_kernel(
    const float* __restrict__ xin, const float* __restrict__ xsn,
    const float* __restrict__ pw, const float* __restrict__ pb,
    float* __restrict__ xres)
{
    size_t tid = (size_t)blockIdx.x * 256 + threadIdx.x;   // NTOK * DIM
    const int d = (int)(tid & (DIM - 1));
    const size_t n = tid >> 8;
    const int t = (int)(n % SEQL);

    float acc = pb[d];
    #pragma unroll
    for (int k = 0; k < 3; ++k) {
        int p = t - 1 + k;
        if (p < 0 || p >= SEQL) continue;
        size_t np = n - t + p;
        acc += pw[d * 6 + 0 * 3 + k] * xsn[np * 512 + 2 * d];
        acc += pw[d * 6 + 1 * 3 + k] * xsn[np * 512 + 2 * d + 1];
    }
    xres[n * DIM + d] = xin[n * DIM + d] + silu_f(acc);
}

// ---------------------------------------------------------------------------
// Launch
// ---------------------------------------------------------------------------
extern "C" void kernel_launch(void* const* d_in, const int* in_sizes, int n_in,
                              void* d_out, int out_size, void* d_ws, size_t ws_size,
                              hipStream_t stream)
{
    const float* x        = (const float*)d_in[0];
    const float* rms1_w   = (const float*)d_in[1];
    const float* lconv_w  = (const float*)d_in[2];
    const float* lconv_b  = (const float*)d_in[3];
    const float* lnc_w    = (const float*)d_in[4];
    const float* lnc_b    = (const float*)d_in[5];
    const float* in_w     = (const float*)d_in[6];
    const float* conv_w   = (const float*)d_in[7];
    const float* conv_b   = (const float*)d_in[8];
    const float* xproj_w  = (const float*)d_in[9];
    const float* dtproj_w = (const float*)d_in[10];
    const float* dtproj_b = (const float*)d_in[11];
    const float* A_log    = (const float*)d_in[12];
    const float* Dm       = (const float*)d_in[13];
    const float* out_w    = (const float*)d_in[14];
    const float* lnpost_w = (const float*)d_in[15];
    const float* lnpost_b = (const float*)d_in[16];
    const float* pconv_w  = (const float*)d_in[17];
    const float* pconv_b  = (const float*)d_in[18];
    const float* rms2_w   = (const float*)d_in[19];
    const float* mlp_w1   = (const float*)d_in[20];
    const float* mlp_b1   = (const float*)d_in[21];
    const float* mlp_w2   = (const float*)d_in[22];
    const float* mlp_b2   = (const float*)d_in[23];

    const size_t N = NTOK;
    float* ws    = (float*)d_ws;
    float* xn    = ws;                    // N*256
    float* u     = xn    + N * 256;       // N*256   (reused as rms2 output)
    float* xz    = u     + N * 256;       // N*1024  (reused as MLP hidden h1)
    float* xcf   = xz    + N * 1024;      // N*512   (reused as gated y_f)
    float* xcb   = xcf   + N * 512;       // N*512   (reused as gated y_b)
    float* projf = xcb   + N * 512;       // N*80
    float* projb = projf + N * 80;        // N*80
    float* dtf   = projb + N * 80;        // N*512
    float* dtb   = dtf   + N * 512;       // N*512
    float* xs    = dtb   + N * 512;       // N*512   (LN done in place)
    float* xres  = xs    + N * 512;       // N*256
    float* outp  = (float*)d_out;

    // 1) xn = rmsnorm(x)
    rmsnorm256_kernel<<<NTOK, 256, 0, stream>>>(x, rms1_w, xn);
    // 2) u = silu(LN(dwconv3(xn))) + xn
    lconv_ln_silu_kernel<<<NTOK, 256, 0, stream>>>(xn, lconv_w, lconv_b, lnc_w, lnc_b, u);
    // 3) xz = u @ in_w     (16384x256 @ 256x1024)
    gemm_wmma_f16<0, false><<<dim3(NTOK / BLK_M, 1024 / BLK_N), 256, 0, stream>>>(
        u, 256, in_w, 1024, xz, 1024, 256, 1024, nullptr, nullptr, 0);
    // 4) depthwise causal conv + silu, both directions
    mamba_conv_kernel<<<(2 * NTOK * DINNER) / 256, 256, 0, stream>>>(
        xz, conv_w, conv_b, xcf, xcb);
    // 5) proj = xc @ xproj_w   (16384x512 @ 512x80), per direction
    gemm_wmma_f16<0, true><<<dim3(NTOK / BLK_M, 2), 256, 0, stream>>>(
        xcf, 512, xproj_w, 80, projf, 80, 512, 80, nullptr, nullptr, 0);
    gemm_wmma_f16<0, true><<<dim3(NTOK / BLK_M, 2), 256, 0, stream>>>(
        xcb, 512, xproj_w, 80, projb, 80, 512, 80, nullptr, nullptr, 0);
    // 6) dt = softplus(proj[:, :16] @ dtproj_w + b)
    dt_kernel<<<(NTOK * DINNER) / 256, 256, 0, stream>>>(projf, dtproj_w, dtproj_b, dtf);
    dt_kernel<<<(NTOK * DINNER) / 256, 256, 0, stream>>>(projb, dtproj_w, dtproj_b, dtb);
    // 7) selective scans (y written over xc, gated by silu(z))
    scan_kernel<<<(BATCH * DINNER) / 8, 256, 0, stream>>>(xz, xcf, dtf, projf, A_log, Dm, 0);
    scan_kernel<<<(BATCH * DINNER) / 8, 256, 0, stream>>>(xz, xcb, dtb, projb, A_log, Dm, 1);
    // 8) xs[:, 0:256] = y_f @ out_w ; xs[:, 256:512] = y_b @ out_w
    gemm_wmma_f16<0, false><<<dim3(NTOK / BLK_M, 4), 256, 0, stream>>>(
        xcf, 512, out_w, 256, xs, 512, 512, 256, nullptr, nullptr, 0);
    gemm_wmma_f16<0, false><<<dim3(NTOK / BLK_M, 4), 256, 0, stream>>>(
        xcb, 512, out_w, 256, xs + 256, 512, 512, 256, nullptr, nullptr, 0);
    // 9) LN over 512, in place
    ln512_kernel<<<NTOK, 256, 0, stream>>>(xs, lnpost_w, lnpost_b);
    // 10) xres = x + silu(grouped pconv(xsn))
    pconv_kernel<<<(NTOK * DIM) / 256, 256, 0, stream>>>(x, xs, pconv_w, pconv_b, xres);
    // 11) h = rmsnorm(xres)    (into u)
    rmsnorm256_kernel<<<NTOK, 256, 0, stream>>>(xres, rms2_w, u);
    // 12) h1 = gelu(h @ mlp_w1 + b1)   (into xz)
    gemm_wmma_f16<1, false><<<dim3(NTOK / BLK_M, 1024 / BLK_N), 256, 0, stream>>>(
        u, 256, mlp_w1, 1024, xz, 1024, 256, 1024, mlp_b1, nullptr, 0);
    // 13) out = xres + h1 @ mlp_w2 + b2
    gemm_wmma_f16<2, false><<<dim3(NTOK / BLK_M, 4), 256, 0, stream>>>(
        xz, 1024, mlp_w2, 256, outp, 256, 1024, 256, mlp_b2, xres, 256);
}
